// LERFField_17291538334450
// MI455X (gfx1250) — matrix operand order repaired and checked by reading, not compiled, not run on hardware
//
#include <hip/hip_runtime.h>
#include <hip/hip_bf16.h>
#include <stdint.h>

typedef __attribute__((ext_vector_type(16))) _Float16 v16h;
typedef __attribute__((ext_vector_type(8)))  float    v8f;

union HalfFrag { uint32_t u[8]; v16h h; };
union Pack8h  { _Float16 h[8]; uint4 u; };

#define LDS_STRIDE 40   // halves/row: 80B rows -> 16B aligned b128 stores, conflict-free frag reads

// ---------------------------------------------------------------------------
// CDNA5 async global->LDS DMA helpers (ASYNCcnt-tracked; CDNA5 ISA)
// ---------------------------------------------------------------------------
__device__ __forceinline__ void async_copy_b128(const _Float16* lds_dst, const _Float16* gsrc) {
  // Generic LDS pointer: ADDR[31:0] is the wave-relative LDS byte address.
  uint32_t lds = (uint32_t)(uintptr_t)lds_dst;
  uint64_t g   = (uint64_t)(uintptr_t)gsrc;
  asm volatile("global_load_async_to_lds_b128 %0, %1, off"
               :: "v"(lds), "v"(g)
               : "memory");
}
__device__ __forceinline__ void wait_async0() {
  asm volatile("s_wait_asynccnt 0x0" ::: "memory");
}

// ---------------------------------------------------------------------------
// Weight prep: f32 [K,M] -> f16 transposed [M,K] (per-launch, tiny)
// ---------------------------------------------------------------------------
__global__ __launch_bounds__(256) void transpose_w_kernel(const float* __restrict__ in,
                                                          _Float16* __restrict__ out,
                                                          int K, int M) {
  const int b = blockIdx.y;
  in  += (size_t)b * K * M;
  out += (size_t)b * K * M;
  int i = blockIdx.x * 256 + threadIdx.x;
  if (i < K * M) {
    int m = i / K, k = i % K;
    out[i] = (_Float16)in[(size_t)k * M + m];
  }
}

// ---------------------------------------------------------------------------
// Hash-grid encoding: one thread per (point, level24).
// Writes X [N, 192] row-major f16 (table0 levels -> cols 0..95, table1 -> 96..191)
// ---------------------------------------------------------------------------
__global__ __launch_bounds__(256) void hash_encode_kernel(
    const float* __restrict__ pos,
    const float4* __restrict__ tab0,
    const float4* __restrict__ tab1,
    _Float16* __restrict__ X, int N) {
  int gid = blockIdx.x * 256 + threadIdx.x;
  if (gid >= N * 24) return;
  int n     = gid / 24;
  int lv24  = gid % 24;
  int which = (lv24 >= 12) ? 1 : 0;
  int lv    = lv24 - which * 12;

  float px = pos[n * 3 + 0], py = pos[n * 3 + 1], pz = pos[n * 3 + 2];
  float mag = sqrtf(px * px + py * py + pz * pz);
  if (mag >= 1.0f) {
    float s = (2.0f - 1.0f / mag) / mag;
    px *= s; py *= s; pz *= s;
  }
  px = (px + 2.0f) * 0.25f;
  py = (py + 2.0f) * 0.25f;
  pz = (pz + 2.0f) * 0.25f;

  float start = which ? 128.0f : 16.0f;
  float endr  = which ? 512.0f : 128.0f;
  float growth = expf((logf(endr) - logf(start)) * (1.0f / 11.0f));
  float res = floorf(start * powf(growth, (float)lv));

  float fx = px * res, fy = py * res, fz = pz * res;
  float x0f = floorf(fx), y0f = floorf(fy), z0f = floorf(fz);
  float wx = fx - x0f, wy = fy - y0f, wz = fz - z0f;
  uint32_t x0 = (uint32_t)x0f, y0 = (uint32_t)y0f, z0 = (uint32_t)z0f;

  const float4* tab = (which ? tab1 : tab0) + (size_t)lv * ((size_t)(1 << 19) * 2);

  float acc[8];
#pragma unroll
  for (int i = 0; i < 8; ++i) acc[i] = 0.0f;

#pragma unroll
  for (int c = 0; c < 8; ++c) {
    uint32_t cx = x0 + ((c >> 2) & 1);
    uint32_t cy = y0 + ((c >> 1) & 1);
    uint32_t cz = z0 + (c & 1);
    uint32_t h  = (cx * 1u) ^ (cy * 2654435761u) ^ (cz * 805459861u);
    uint32_t idx = h & 524287u;   // T-1, T = 2^19
    float wgt = ((c & 4) ? wx : 1.0f - wx) *
                ((c & 2) ? wy : 1.0f - wy) *
                ((c & 1) ? wz : 1.0f - wz);
    float4 f0 = tab[(size_t)idx * 2 + 0];
    float4 f1 = tab[(size_t)idx * 2 + 1];
    acc[0] += wgt * f0.x; acc[1] += wgt * f0.y; acc[2] += wgt * f0.z; acc[3] += wgt * f0.w;
    acc[4] += wgt * f1.x; acc[5] += wgt * f1.y; acc[6] += wgt * f1.z; acc[7] += wgt * f1.w;
  }

  Pack8h p;
#pragma unroll
  for (int i = 0; i < 8; ++i) p.h[i] = (_Float16)acc[i];
  *(uint4*)(X + (size_t)n * 192 + which * 96 + lv * 8) = p.u;
}

// ---------------------------------------------------------------------------
// Tiled WMMA GEMM:  C[N,M] = act( A[N,K] @ Bt[M,K]^T )
// block tile 128x64, 8 waves (4x2) each computing 32x32 via 2x2 WMMA 16x16x32 f16.
// Double-buffered LDS; tiles staged with global_load_async_to_lds_b128 and
// overlapped with WMMA compute of the previous tile.
// ---------------------------------------------------------------------------
template <bool RELU, bool OUTF32>
__global__ __launch_bounds__(256) void wmma_gemm_kernel(
    const _Float16* __restrict__ A, int lda,
    const _Float16* __restrict__ Bt, int ldbt,   // transposed weights [M,K]
    void* __restrict__ C, int ldc, int colOff, int K) {
  __shared__ _Float16 sA[2][128 * LDS_STRIDE];  // A tile [128][32], row-major
  __shared__ _Float16 sB[2][64  * LDS_STRIDE];  // Bt tile [64][32], n-major (K contiguous)

  const int tid  = threadIdx.x;
  const int wave = tid >> 5;
  const int lane = tid & 31;
  const int wr   = wave >> 1;  // 0..3 : 32-row slab within 128
  const int wc   = wave & 1;   // 0..1 : 32-col slab within 64
  const int lm   = lane & 15;
  const int lhi  = lane >> 4;

  const size_t rowBase = (size_t)blockIdx.y * 128;
  const int    colBase = blockIdx.x * 64;

  // per-thread staging chunk coordinates (16B chunks)
  const int ar0 = tid >> 2,         as0 = tid & 3;          // A chunk 0
  const int ar1 = (tid + 256) >> 2, as1 = tid & 3;          // A chunk 1
  const int bn  = tid >> 2,         bs  = tid & 3;          // B chunk

  v8f acc[2][2] = {};

  const int nk = K >> 5;

  // prologue: stage tile 0
  {
    async_copy_b128(&sA[0][ar0 * LDS_STRIDE + as0 * 8], A + (rowBase + ar0) * lda + as0 * 8);
    async_copy_b128(&sA[0][ar1 * LDS_STRIDE + as1 * 8], A + (rowBase + ar1) * lda + as1 * 8);
    async_copy_b128(&sB[0][bn  * LDS_STRIDE + bs  * 8], Bt + (size_t)(colBase + bn) * ldbt + bs * 8);
  }

  for (int i = 0; i < nk; ++i) {
    wait_async0();        // this wave's tile-i DMA done
    __syncthreads();      // all waves' tile-i DMA done; tile-(i-1) reads finished

    if (i + 1 < nk) {     // overlap tile-(i+1) DMA with tile-i compute
      const int k0n = (i + 1) << 5;
      const int b   = (i + 1) & 1;
      async_copy_b128(&sA[b][ar0 * LDS_STRIDE + as0 * 8], A + (rowBase + ar0) * lda + k0n + as0 * 8);
      async_copy_b128(&sA[b][ar1 * LDS_STRIDE + as1 * 8], A + (rowBase + ar1) * lda + k0n + as1 * 8);
      async_copy_b128(&sB[b][bn  * LDS_STRIDE + bs  * 8], Bt + (size_t)(colBase + bn) * ldbt + k0n + bs * 8);
    }

    const _Float16* __restrict__ cA = sA[i & 1];
    const _Float16* __restrict__ cB = sB[i & 1];

    // ---- fragments per CDNA5 VGPR layouts
    HalfFrag a[2], b[2];
#pragma unroll
    for (int t = 0; t < 2; ++t) {
      const int m  = wr * 32 + t * 16 + lm;
      const int kb = lhi * 8;  // A: lanes 0-15 -> K {0..7,16..23}; lanes 16-31 -> {8..15,24..31}
      const _Float16* base = &cA[m * LDS_STRIDE];
#pragma unroll
      for (int r = 0; r < 4; ++r) a[t].u[r]     = *(const uint32_t*)&base[kb + 2 * r];
#pragma unroll
      for (int r = 0; r < 4; ++r) a[t].u[4 + r] = *(const uint32_t*)&base[16 + kb + 2 * r];
    }
#pragma unroll
    for (int t = 0; t < 2; ++t) {
      const int n  = wc * 32 + t * 16 + lm;
      const int kb = lhi * 8;  // B: lanes 0-15 -> K {0..7,16..23}; lanes 16-31 -> {8..15,24..31}
      const _Float16* base = &cB[n * LDS_STRIDE];
#pragma unroll
      for (int r = 0; r < 4; ++r) b[t].u[r]     = *(const uint32_t*)&base[kb + 2 * r];
#pragma unroll
      for (int r = 0; r < 4; ++r) b[t].u[4 + r] = *(const uint32_t*)&base[16 + kb + 2 * r];
    }

#pragma unroll
    for (int t = 0; t < 2; ++t)
#pragma unroll
      for (int u = 0; u < 2; ++u)
        acc[t][u] = __builtin_amdgcn_wmma_f32_16x16x32_f16(
            false, a[t].h, false, b[u].h, (short)0, acc[t][u], false, false);
  }

  // ---- epilogue: C layout — N = lane&15, M = 8*(lane>=16) + vgpr
#pragma unroll
  for (int t = 0; t < 2; ++t) {
#pragma unroll
    for (int u = 0; u < 2; ++u) {
#pragma unroll
      for (int r = 0; r < 8; ++r) {
        const int m  = lhi * 8 + r;
        const size_t row = rowBase + wr * 32 + t * 16 + m;
        const int    col = colBase + wc * 32 + u * 16 + lm;
        float v = acc[t][u][r];
        if (RELU) v = fmaxf(v, 0.0f);
        if (OUTF32) ((float*)C)[row * (size_t)ldc + colOff + col] = v;
        else        ((_Float16*)C)[row * (size_t)ldc + colOff + col] = (_Float16)v;
      }
    }
  }
}

// ---------------------------------------------------------------------------
// In-place: clip = (clip + 1e-8) / max(||clip||, 1e-4), cols 0..511 of [N,896]
// ---------------------------------------------------------------------------
__global__ __launch_bounds__(256) void clip_normalize_kernel(float* __restrict__ out) {
  __shared__ float red[256];
  const size_t base = (size_t)blockIdx.x * 896;
  float v0 = out[base + threadIdx.x]       + 1e-8f;
  float v1 = out[base + threadIdx.x + 256] + 1e-8f;
  red[threadIdx.x] = v0 * v0 + v1 * v1;
  __syncthreads();
#pragma unroll
  for (int off = 128; off > 0; off >>= 1) {
    if (threadIdx.x < off) red[threadIdx.x] += red[threadIdx.x + off];
    __syncthreads();
  }
  const float inv = 1.0f / fmaxf(sqrtf(red[0]), 1e-4f);
  out[base + threadIdx.x]       = v0 * inv;
  out[base + threadIdx.x + 256] = v1 * inv;
}

// ---------------------------------------------------------------------------
extern "C" void kernel_launch(void* const* d_in, const int* in_sizes, int n_in,
                              void* d_out, int out_size, void* d_ws, size_t ws_size,
                              hipStream_t stream) {
  const float* positions  = (const float*)d_in[0];
  const float* table0     = (const float*)d_in[1];
  const float* table1     = (const float*)d_in[2];
  const float* clip_w_in  = (const float*)d_in[3];
  const float* clip_w_h   = (const float*)d_in[4];
  const float* clip_w_out = (const float*)d_in[5];
  const float* dino_w_in  = (const float*)d_in[6];
  const float* dino_w_out = (const float*)d_in[7];
  float* out = (float*)d_out;

  const int N = in_sizes[0] / 3;  // 131072
  dim3 blk(256);

  // workspace layout (transposed f16 weights + activations)
  char* wsp = (char*)d_ws;
  _Float16* wf      = (_Float16*)wsp;                 // 524288 halves = 1 MB
  _Float16* wcT_in  = wf;                             // [256,192]
  _Float16* wcT_h   = wf + 49152;                     // 3 x [256,256]
  _Float16* wcT_out = wf + 49152 + 196608;            // [512,256]
  _Float16* wdT_in  = wcT_out + 131072;               // [256,192]
  _Float16* wdT_out = wdT_in + 49152;                 // [384,256]
  _Float16* Xf      = (_Float16*)(wsp + (1 << 20));
  _Float16* ping    = Xf + (size_t)N * 192;
  _Float16* pong    = ping + (size_t)N * 256;

  // 1) weights -> transposed f16 [M,K]
  transpose_w_kernel<<<dim3((192 * 256 + 255) / 256, 1), blk, 0, stream>>>(clip_w_in,  wcT_in,  192, 256);
  transpose_w_kernel<<<dim3((256 * 256 + 255) / 256, 3), blk, 0, stream>>>(clip_w_h,   wcT_h,   256, 256);
  transpose_w_kernel<<<dim3((256 * 512 + 255) / 256, 1), blk, 0, stream>>>(clip_w_out, wcT_out, 256, 512);
  transpose_w_kernel<<<dim3((192 * 256 + 255) / 256, 1), blk, 0, stream>>>(dino_w_in,  wdT_in,  192, 256);
  transpose_w_kernel<<<dim3((256 * 384 + 255) / 256, 1), blk, 0, stream>>>(dino_w_out, wdT_out, 256, 384);

  // 2) hash encode -> X f16 [N,192]
  hash_encode_kernel<<<(N * 24 + 255) / 256, blk, 0, stream>>>(
      positions, (const float4*)table0, (const float4*)table1, Xf, N);

  // 3) MLP chain via WMMA GEMMs (B passed transposed, ldbt = K)
  dim3 g256(256 / 64, N / 128);
  wmma_gemm_kernel<true,  false><<<g256, blk, 0, stream>>>(Xf,   192, wcT_in,           192, ping, 256, 0,   192);
  wmma_gemm_kernel<true,  false><<<g256, blk, 0, stream>>>(ping, 256, wcT_h,            256, pong, 256, 0,   256);
  wmma_gemm_kernel<true,  false><<<g256, blk, 0, stream>>>(pong, 256, wcT_h +  65536,   256, ping, 256, 0,   256);
  wmma_gemm_kernel<true,  false><<<g256, blk, 0, stream>>>(ping, 256, wcT_h + 131072,   256, pong, 256, 0,   256);
  wmma_gemm_kernel<false, true ><<<dim3(512 / 64, N / 128), blk, 0, stream>>>(
      pong, 256, wcT_out, 256, out, 896, 0, 256);
  wmma_gemm_kernel<true,  false><<<g256, blk, 0, stream>>>(Xf,   192, wdT_in,           192, ping, 256, 0,   192);
  wmma_gemm_kernel<false, true ><<<dim3(384 / 64, N / 128), blk, 0, stream>>>(
      ping, 256, wdT_out, 256, out, 896, 512, 256);

  // 4) normalize clip half in place
  clip_normalize_kernel<<<N, blk, 0, stream>>>(out);

  (void)n_in; (void)out_size; (void)ws_size;
}
